// localfeaFromImage_24618752541274
// MI455X (gfx1250) — matrix operand order repaired and checked by reading, not compiled, not run on hardware
//
#include <hip/hip_runtime.h>
#include <hip/hip_bf16.h>

typedef __attribute__((ext_vector_type(16))) _Float16 v16h;
typedef __attribute__((ext_vector_type(8)))  float    v8f;

#define KP    68
#define BATCH 128
#define HH    224
#define EPSB  1e-5f

// Compile-time im2col offset tables (folded to integer literals after unroll).
// conv1: K index -> ic*50176 + ky*224 + kx   (3 ic, 6x6 kernel, x is 224x224)
static constexpr int im2col1(int ka) {
  return (ka >= 108) ? 0
       : ((ka / 36) * 50176 + (((ka % 36) / 6) * 224) + ((ka % 36) % 6));
}
// conv2: K index -> ic*49 + ky*7 + kx        (16 ic, 3x3 kernel, maps are 7x7)
static constexpr int im2col2(int ka) {
  return (ka >= 144) ? 0
       : ((ka / 9) * 49 + (((ka % 9) / 3) * 7) + ((ka % 9) % 3));
}

// ---------------------------------------------------------------------------
// Global feature: mean over 7x7 of relu(features)  [B,2208,7,7] -> [B,2208]
// ---------------------------------------------------------------------------
__global__ void k_gfeat(const float* __restrict__ feat, float* __restrict__ gfeat, int total) {
  int i = blockIdx.x * blockDim.x + threadIdx.x;
  if (i >= total) return;
  const float* p = feat + (size_t)i * 49;
  float s = 0.f;
#pragma unroll
  for (int j = 0; j < 49; ++j) { float v = p[j]; s += v > 0.f ? v : 0.f; }
  gfeat[i] = s * (1.0f / 49.0f);
}

// ---------------------------------------------------------------------------
// conv1: per-k implicit GEMM  M=B*169, N=16(oc), K=108(pad 128), f16 WMMA
// 4 waves / block, one 16x16 output tile per wave. Branchless literal gathers.
// ---------------------------------------------------------------------------
__global__ __launch_bounds__(128)
void k_conv1(const float* __restrict__ x, const float* __restrict__ points,
             const float* __restrict__ c1w, const float* __restrict__ c1b,
             _Float16* __restrict__ y1, float* __restrict__ stat1)
{
  const int tilesPerK = (BATCH * 169) / 16;  // 1352
  int gt   = blockIdx.x * 4 + (threadIdx.x >> 5);
  int k    = gt / tilesPerK;
  int tile = gt - k * tilesPerK;
  int l    = threadIdx.x & 31;
  int half = l >> 4;
  int lm   = l & 15;
  int m0   = tile * 16;

  // A-row (im2col) owned by this lane
  int r   = m0 + lm;
  int b   = r / 169;
  int pos = r - b * 169;
  int oy  = pos / 13;
  int ox  = pos - oy * 13;
  float p0 = points[b * (2 * KP) + 2 * k];
  float p1 = points[b * (2 * KP) + 2 * k + 1];
  int s0 = (int)((p0 + 0.5f) * 224.0f) - 15;
  s0 = s0 < 0 ? 0 : (s0 > 193 ? 193 : s0);
  int s1 = (int)((p1 + 0.5f) * 224.0f) - 15;
  s1 = s1 < 0 ? 0 : (s1 > 193 ? 193 : s1);
  const float* xrow = x + (size_t)b * 3 * 224 * 224
                        + (s0 + oy * 2) * 224 + (s1 + ox * 2);

  // fold lane's half-offset into the weight pointer so indices are literals
  const float* wrow = c1w + (size_t)k * 16 * 108 + lm * 108 + (half << 3);

  v8f acc = {};
#pragma unroll
  for (int step = 0; step < 4; ++step) {
    v16h af, bf;
#pragma unroll
    for (int j = 0; j < 16; ++j) {
      const int ka0 = step * 32 + (j & 7) + ((j & 8) << 1);  // literal
      const int o0 = im2col1(ka0);                            // literal
      const int o1 = im2col1(ka0 + 8);                        // literal
      const bool v0 = ka0 < 108, v1 = (ka0 + 8) < 108;        // literals
      int   off   = half ? o1 : o0;       // one cndmask of two literals
      bool  valid = half ? v1 : v0;
      float av = xrow[off];
      af[j] = (_Float16)(valid ? av : 0.0f);

      const int kb0 = step * 32 + (j & 7) + ((j & 8) << 1);  // literal
      bool bval = half ? (kb0 + 8 < 108) : (kb0 < 108);       // literal bools
      float bv = wrow[bval ? kb0 : 0];
      bf[j] = (_Float16)(bval ? bv : 0.0f);
    }
    acc = __builtin_amdgcn_wmma_f32_16x16x32_f16(false, af, false, bf, (short)0, acc, false, false);
  }

  float bias = c1b[k * 16 + lm];
  float s = 0.f, s2 = 0.f;
#pragma unroll
  for (int v = 0; v < 8; ++v) {
    int m  = m0 + v + 8 * half;
    int bb = m / 169;
    int pp = m - bb * 169;
    float val = acc[v] + bias;
    y1[(((size_t)k * BATCH + bb) * 16 + lm) * 169 + pp] = (_Float16)val;
    s  += val;
    s2 += val * val;
  }
  s  += __shfl_xor(s, 16, 32);
  s2 += __shfl_xor(s2, 16, 32);
  if (half == 0) {
    atomicAdd(&stat1[(k * 16 + lm) * 2 + 0], s);
    atomicAdd(&stat1[(k * 16 + lm) * 2 + 1], s2);
  }
}

// ---------------------------------------------------------------------------
// BN(2d, training stats) + relu + ceil maxpool 13->7 ;  y1 -> p1 (f16)
// ---------------------------------------------------------------------------
__global__ void k_bnpool1(const _Float16* __restrict__ y1, const float* __restrict__ stat1,
                          const float* __restrict__ g, const float* __restrict__ be,
                          _Float16* __restrict__ p1, int total)
{
  int i = blockIdx.x * blockDim.x + threadIdx.x;
  if (i >= total) return;
  int px = i % 7;
  int t  = i / 7;
  int py = t % 7;  t /= 7;
  int oc = t & 15; t >>= 4;
  int b  = t & 127;
  int k  = t >> 7;
  int sidx = k * 16 + oc;
  const float cnt = 128.0f * 169.0f;
  float mean = stat1[sidx * 2] / cnt;
  float var  = stat1[sidx * 2 + 1] / cnt - mean * mean;
  float inv  = rsqrtf(var + EPSB);
  float gg = g[sidx], bb = be[sidx];
  const _Float16* src = y1 + (((size_t)k * BATCH + b) * 16 + oc) * 169;
  float mx = -1e30f;
#pragma unroll
  for (int dy = 0; dy < 2; ++dy) {
    int yy = 2 * py + dy;
    int yyc = yy < 13 ? yy : 12;
#pragma unroll
    for (int dx = 0; dx < 2; ++dx) {
      int xx = 2 * px + dx;
      int xxc = xx < 13 ? xx : 12;
      float v = (float)src[yyc * 13 + xxc];
      v = (v - mean) * inv * gg + bb;
      mx = fmaxf(mx, (yy < 13 && xx < 13) ? v : -1e30f);
    }
  }
  p1[i] = (_Float16)fmaxf(mx, 0.0f);  // relu commutes with max
}

// ---------------------------------------------------------------------------
// conv2: per-k implicit GEMM  M=B*25, N=16, K=144(pad 160), f16 WMMA
// ---------------------------------------------------------------------------
__global__ __launch_bounds__(128)
void k_conv2(const _Float16* __restrict__ p1, const float* __restrict__ c2w,
             const float* __restrict__ c2b, _Float16* __restrict__ y2,
             float* __restrict__ stat2)
{
  const int tilesPerK = (BATCH * 25) / 16;  // 200
  int gt   = blockIdx.x * 4 + (threadIdx.x >> 5);
  int k    = gt / tilesPerK;
  int tile = gt - k * tilesPerK;
  int l    = threadIdx.x & 31;
  int half = l >> 4;
  int lm   = l & 15;
  int m0   = tile * 16;

  int r   = m0 + lm;
  int b   = r / 25;
  int pos = r - b * 25;
  int oy  = pos / 5;
  int ox  = pos - oy * 5;
  const _Float16* srow = p1 + ((size_t)k * BATCH + b) * 16 * 49 + oy * 7 + ox;
  const float* wrow = c2w + (size_t)k * 16 * 144 + lm * 144 + (half << 3);

  v8f acc = {};
#pragma unroll
  for (int step = 0; step < 5; ++step) {
    v16h af, bf;
#pragma unroll
    for (int j = 0; j < 16; ++j) {
      const int ka0 = step * 32 + (j & 7) + ((j & 8) << 1);   // literal
      const int o0 = im2col2(ka0);                             // literal
      const int o1 = im2col2(ka0 + 8);                         // literal
      const bool v0 = ka0 < 144, v1 = (ka0 + 8) < 144;
      int  off   = half ? o1 : o0;
      bool valid = half ? v1 : v0;
      float av = (float)srow[off];
      af[j] = (_Float16)(valid ? av : 0.0f);

      const int kb0 = step * 32 + (j & 7) + ((j & 8) << 1);   // literal
      bool bval = half ? (kb0 + 8 < 144) : (kb0 < 144);
      float bv = wrow[bval ? kb0 : 0];
      bf[j] = (_Float16)(bval ? bv : 0.0f);
    }
    acc = __builtin_amdgcn_wmma_f32_16x16x32_f16(false, af, false, bf, (short)0, acc, false, false);
  }

  float bias = c2b[k * 16 + lm];
  float s = 0.f, s2 = 0.f;
#pragma unroll
  for (int v = 0; v < 8; ++v) {
    int m  = m0 + v + 8 * half;
    int bb = m / 25;
    int pp = m - bb * 25;
    float val = acc[v] + bias;
    y2[(((size_t)k * BATCH + bb) * 16 + lm) * 25 + pp] = (_Float16)val;
    s  += val;
    s2 += val * val;
  }
  s  += __shfl_xor(s, 16, 32);
  s2 += __shfl_xor(s2, 16, 32);
  if (half == 0) {
    atomicAdd(&stat2[(k * 16 + lm) * 2 + 0], s);
    atomicAdd(&stat2[(k * 16 + lm) * 2 + 1], s2);
  }
}

// ---------------------------------------------------------------------------
// BN + relu + ceil avgpool 5->3 (divide by clipped count) ; y2 -> f144 (f16)
// ---------------------------------------------------------------------------
__global__ void k_bnpool2(const _Float16* __restrict__ y2, const float* __restrict__ stat2,
                          const float* __restrict__ g, const float* __restrict__ be,
                          _Float16* __restrict__ f144, int total)
{
  int i = blockIdx.x * blockDim.x + threadIdx.x;
  if (i >= total) return;
  int px = i % 3;
  int t  = i / 3;
  int py = t % 3;  t /= 3;
  int oc = t & 15; t >>= 4;
  int b  = t & 127;
  int k  = t >> 7;
  int sidx = k * 16 + oc;
  const float cnt = 128.0f * 25.0f;
  float mean = stat2[sidx * 2] / cnt;
  float var  = stat2[sidx * 2 + 1] / cnt - mean * mean;
  float inv  = rsqrtf(var + EPSB);
  float gg = g[sidx], bb = be[sidx];
  const _Float16* src = y2 + (((size_t)k * BATCH + b) * 16 + oc) * 25;
  float sum = 0.f;
  float c = 0.f;
#pragma unroll
  for (int dy = 0; dy < 2; ++dy) {
    int yy = 2 * py + dy;
    int yyc = yy < 5 ? yy : 4;
#pragma unroll
    for (int dx = 0; dx < 2; ++dx) {
      int xx = 2 * px + dx;
      int xxc = xx < 5 ? xx : 4;
      float v = ((float)src[yyc * 5 + xxc] - mean) * inv * gg + bb;
      v = v > 0.f ? v : 0.f;
      bool ok = (yy < 5 && xx < 5);
      sum += ok ? v : 0.f;
      c   += ok ? 1.f : 0.f;
    }
  }
  f144[i] = (_Float16)(sum / c);
}

// ---------------------------------------------------------------------------
// FC1: per-k GEMM  M=128, N=128, K=144(pad 160), f16 WMMA -> z1 (f32, +bias)
// ---------------------------------------------------------------------------
__global__ __launch_bounds__(128)
void k_linear1(const _Float16* __restrict__ f144, const float* __restrict__ l1w,
               const float* __restrict__ l1b, float* __restrict__ z1)
{
  int gt = blockIdx.x * 4 + (threadIdx.x >> 5);   // KP*8*8 tiles total
  int k  = gt / 64;
  int t  = gt - k * 64;
  int mt = t >> 3, nt = t & 7;
  int l = threadIdx.x & 31, half = l >> 4, lm = l & 15;
  int m0 = mt * 16, n0 = nt * 16;
  const _Float16* arow = f144 + ((size_t)k * BATCH + (m0 + lm)) * 144 + (half << 3);
  const float*    wrow = l1w + (size_t)k * 128 * 144 + (size_t)(n0 + lm) * 144 + (half << 3);

  v8f acc = {};
#pragma unroll
  for (int step = 0; step < 5; ++step) {
    v16h af, bf;
#pragma unroll
    for (int j = 0; j < 16; ++j) {
      const int ka0 = step * 32 + (j & 7) + ((j & 8) << 1);   // literal
      bool aval = half ? (ka0 + 8 < 144) : (ka0 < 144);
      _Float16 av = arow[aval ? ka0 : 0];
      af[j] = aval ? av : (_Float16)0.0f;

      const int kb0 = step * 32 + (j & 7) + ((j & 8) << 1);   // literal
      bool bval = half ? (kb0 + 8 < 144) : (kb0 < 144);
      float bv = wrow[bval ? kb0 : 0];
      bf[j] = (_Float16)(bval ? bv : 0.0f);
    }
    acc = __builtin_amdgcn_wmma_f32_16x16x32_f16(false, af, false, bf, (short)0, acc, false, false);
  }
  int n = n0 + lm;
  float bias = l1b[k * 128 + n];
#pragma unroll
  for (int v = 0; v < 8; ++v) {
    int m = m0 + v + 8 * half;
    z1[((size_t)k * BATCH + m) * 128 + n] = acc[v] + bias;
  }
}

// BN1d stats over B=128 (no atomics needed)
__global__ void k_stats3(const float* __restrict__ z1, float* __restrict__ stat3, int total) {
  int i = blockIdx.x * blockDim.x + threadIdx.x;
  if (i >= total) return;          // total = KP*128
  int ch = i & 127, k = i >> 7;
  float s = 0.f, s2 = 0.f;
  for (int b = 0; b < BATCH; ++b) {
    float v = z1[((size_t)k * BATCH + b) * 128 + ch];
    s += v; s2 += v * v;
  }
  stat3[i * 2] = s; stat3[i * 2 + 1] = s2;
}

__global__ void k_bn3(const float* __restrict__ z1, const float* __restrict__ stat3,
                      const float* __restrict__ g, const float* __restrict__ be,
                      _Float16* __restrict__ a1, int total)
{
  int i = blockIdx.x * blockDim.x + threadIdx.x;
  if (i >= total) return;
  int ch = i & 127;
  int t  = i >> 7;
  int k  = t >> 7;
  int sidx = k * 128 + ch;
  float mean = stat3[sidx * 2] * (1.0f / 128.0f);
  float var  = stat3[sidx * 2 + 1] * (1.0f / 128.0f) - mean * mean;
  float inv  = rsqrtf(var + EPSB);
  float v = (z1[i] - mean) * inv * g[sidx] + be[sidx];
  a1[i] = (_Float16)(v > 0.f ? v : 0.f);
}

// ---------------------------------------------------------------------------
// FC2: per-k GEMM  M=128, N=136(pad 144), K=128, f16 WMMA -> lf (f32, +bias)
// ---------------------------------------------------------------------------
__global__ __launch_bounds__(128)
void k_linear2(const _Float16* __restrict__ a1, const float* __restrict__ l2w,
               const float* __restrict__ l2b, float* __restrict__ lf)
{
  int gt = blockIdx.x * 4 + (threadIdx.x >> 5);   // KP*8*9 tiles total
  int k  = gt / 72;
  int t  = gt - k * 72;
  int mt = t / 9, nt = t - mt * 9;
  int l = threadIdx.x & 31, half = l >> 4, lm = l & 15;
  int m0 = mt * 16, n0 = nt * 16;
  int n  = n0 + lm;
  bool nvalid = (n < 136);
  const _Float16* arow = a1 + ((size_t)k * BATCH + (m0 + lm)) * 128 + (half << 3);
  const float*    wrow = l2w + (size_t)k * 136 * 128 + (size_t)(nvalid ? n : 0) * 128 + (half << 3);

  v8f acc = {};
#pragma unroll
  for (int step = 0; step < 4; ++step) {
    v16h af, bf;
#pragma unroll
    for (int j = 0; j < 16; ++j) {
      const int ka0 = step * 32 + (j & 7) + ((j & 8) << 1);  // literal, always valid
      af[j] = arow[ka0];
      const int kb0 = step * 32 + (j & 7) + ((j & 8) << 1);  // literal, always valid
      float bv = wrow[kb0];
      bf[j] = (_Float16)(nvalid ? bv : 0.0f);
    }
    acc = __builtin_amdgcn_wmma_f32_16x16x32_f16(false, af, false, bf, (short)0, acc, false, false);
  }
  if (nvalid) {
    float bias = l2b[k * 136 + n];
#pragma unroll
    for (int v = 0; v < 8; ++v) {
      int m = m0 + v + 8 * half;
      lf[((size_t)k * BATCH + m) * 136 + n] = acc[v] + bias;
    }
  }
}

// ---------------------------------------------------------------------------
// Final projection: out[b, 2k+o] = pre_b + <fea, pre_w[k,o]> , fea = [gfeat|lf]
// ---------------------------------------------------------------------------
__global__ __launch_bounds__(128)
void k_final(const float* __restrict__ gfeat, const float* __restrict__ lf,
             const float* __restrict__ prw, const float* __restrict__ prb,
             float* __restrict__ out)
{
  int k = blockIdx.x >> 7;   // grid = KP*BATCH
  int b = blockIdx.x & 127;
  int tid = threadIdx.x;
  const float* w  = prw + (size_t)k * 2 * 2344;
  const float* gb = gfeat + (size_t)b * 2208;
  const float* lb = lf + ((size_t)k * BATCH + b) * 136;
  float a0 = 0.f, a1 = 0.f;
  for (int f = tid; f < 2344; f += 128) {
    float v = (f < 2208) ? gb[f] : lb[f - 2208];
    a0 += v * w[f];
    a1 += v * w[2344 + f];
  }
  __shared__ float r0[128], r1[128];
  r0[tid] = a0; r1[tid] = a1;
  __syncthreads();
  for (int s = 64; s > 0; s >>= 1) {
    if (tid < s) { r0[tid] += r0[tid + s]; r1[tid] += r1[tid + s]; }
    __syncthreads();
  }
  if (tid == 0) {
    out[b * 136 + 2 * k]     = r0[0] + prb[k * 2];
    out[b * 136 + 2 * k + 1] = r1[0] + prb[k * 2 + 1];
  }
}

// ---------------------------------------------------------------------------
extern "C" void kernel_launch(void* const* d_in, const int* in_sizes, int n_in,
                              void* d_out, int out_size, void* d_ws, size_t ws_size,
                              hipStream_t stream)
{
  const float* x      = (const float*)d_in[0];
  const float* points = (const float*)d_in[1];
  const float* feats  = (const float*)d_in[2];
  const float* c1w = (const float*)d_in[3];
  const float* c1b = (const float*)d_in[4];
  const float* g1  = (const float*)d_in[5];
  const float* b1  = (const float*)d_in[6];
  const float* c2w = (const float*)d_in[7];
  const float* c2b = (const float*)d_in[8];
  const float* g2  = (const float*)d_in[9];
  const float* b2  = (const float*)d_in[10];
  const float* l1w = (const float*)d_in[11];
  const float* l1b = (const float*)d_in[12];
  const float* g3  = (const float*)d_in[13];
  const float* b3  = (const float*)d_in[14];
  const float* l2w = (const float*)d_in[15];
  const float* l2b = (const float*)d_in[16];
  const float* prw = (const float*)d_in[17];
  const float* prb = (const float*)d_in[18];
  float* out = (float*)d_out;

  char* wsp = (char*)d_ws;
  size_t off = 0;
  auto walloc = [&](size_t bytes) -> void* {
    void* p = wsp + off;
    off = (off + bytes + 255) & ~(size_t)255;
    return p;
  };
  float*    gfeat = (float*)   walloc((size_t)BATCH * 2208 * 4);
  float*    stat1 = (float*)   walloc((size_t)KP * 16 * 2 * 4);
  float*    stat2 = (float*)   walloc((size_t)KP * 16 * 2 * 4);
  float*    stat3 = (float*)   walloc((size_t)KP * 128 * 2 * 4);
  _Float16* y1    = (_Float16*)walloc((size_t)KP * BATCH * 16 * 169 * 2);
  _Float16* p1    = (_Float16*)walloc((size_t)KP * BATCH * 16 * 49 * 2);
  _Float16* y2    = (_Float16*)walloc((size_t)KP * BATCH * 16 * 25 * 2);
  _Float16* f144  = (_Float16*)walloc((size_t)KP * BATCH * 144 * 2);
  float*    z1    = (float*)   walloc((size_t)KP * BATCH * 128 * 4);
  _Float16* a1    = (_Float16*)walloc((size_t)KP * BATCH * 128 * 2);
  float*    lf    = (float*)   walloc((size_t)KP * BATCH * 136 * 4);

  hipMemsetAsync(stat1, 0, (size_t)KP * 16 * 2 * 4, stream);
  hipMemsetAsync(stat2, 0, (size_t)KP * 16 * 2 * 4, stream);

  { int total = BATCH * 2208;
    k_gfeat<<<(total + 255) / 256, 256, 0, stream>>>(feats, gfeat, total); }

  // KP*1352 tiles, 4 waves (tiles) per 128-thread block
  k_conv1<<<KP * 1352 / 4, 128, 0, stream>>>(x, points, c1w, c1b, y1, stat1);

  { int total = KP * BATCH * 16 * 49;
    k_bnpool1<<<(total + 255) / 256, 256, 0, stream>>>(y1, stat1, g1, b1, p1, total); }

  k_conv2<<<KP * 200 / 4, 128, 0, stream>>>(p1, c2w, c2b, y2, stat2);

  { int total = KP * BATCH * 144;
    k_bnpool2<<<(total + 255) / 256, 256, 0, stream>>>(y2, stat2, g2, b2, f144, total); }

  k_linear1<<<KP * 64 / 4, 128, 0, stream>>>(f144, l1w, l1b, z1);

  { int total = KP * 128;
    k_stats3<<<(total + 255) / 256, 256, 0, stream>>>(z1, stat3, total); }

  { int total = KP * BATCH * 128;
    k_bn3<<<(total + 255) / 256, 256, 0, stream>>>(z1, stat3, g3, b3, a1, total); }

  k_linear2<<<KP * 72 / 4, 128, 0, stream>>>(a1, l2w, l2b, lf);

  k_final<<<KP * BATCH, 128, 0, stream>>>(gfeat, lf, prw, prb, out);
}
